// BClassifier_43782896615750
// MI455X (gfx1250) — matrix operand order repaired and checked by reading, not compiled, not run on hardware
//
#include <hip/hip_runtime.h>
#include <hip/hip_bf16.h>
#include <math.h>

#define BATCH 4
#define INST  40000
#define DIM   512
#define HID   128
#define OC    2
#define NB    100            // partial-B blocks per batch
#define CH    (INST/NB)      // 400 instances per partial block

typedef __attribute__((ext_vector_type(16))) _Float16 v16h;
typedef __attribute__((ext_vector_type(8)))  _Float16 v8h;
typedef __attribute__((ext_vector_type(4)))  _Float16 v4h;
typedef __attribute__((ext_vector_type(8)))  float    v8f;

// ---------------------------------------------------------------------------
// WMMA operand loaders (CDNA5 wave32 layouts, 16-bit data)
// A (16x32, MxK): lane%16 = M row; K = kb + 8*(lane/16) + {0..7} and {16..23}
// B (32x16, KxN): lane%16 = N col; K = kb + 8*(lane/16) + {0..7} and {16..23}
//   (weights stored transposed: wT[n][k], so B-load is row-major per lane)
// ---------------------------------------------------------------------------
__device__ __forceinline__ v16h load_opA_lds(const _Float16* base, int stride,
                                             int lane, int kb) {
    int row  = lane & 15;
    int koff = (lane >> 4) * 8;
    const _Float16* p = base + row * stride + kb + koff;
    union { v16h v; v8h h[2]; } u;
    u.h[0] = *(const v8h*)(p);
    u.h[1] = *(const v8h*)(p + 16);
    return u.v;
}

__device__ __forceinline__ v16h load_opB_gl(const _Float16* base, int stride,
                                            int lane, int n0, int kb) {
    int n    = n0 + (lane & 15);
    int koff = (lane >> 4) * 8;
    const _Float16* p = base + (size_t)n * stride + kb + koff;
    union { v16h v; v8h h[2]; } u;
    u.h[0] = *(const v8h*)(p);
    u.h[1] = *(const v8h*)(p + 16);
    return u.v;
}

// ---------------------------------------------------------------------------
// K1: argmax of c along instances -> idx[b*OC+o]  (first max on ties)
// ---------------------------------------------------------------------------
__global__ void k_argmax(const float* __restrict__ c, int* __restrict__ idx) {
    int pair = blockIdx.x;            // 0..7
    int b = pair / OC, o = pair % OC;
    int t = threadIdx.x;
    float best = -INFINITY; int bi = INST;
    for (int n = t; n < INST; n += 256) {
        float v = c[((size_t)b * INST + n) * OC + o];
        if (v > best || (v == best && n < bi)) { best = v; bi = n; }
    }
    __shared__ float sv[256]; __shared__ int si[256];
    sv[t] = best; si[t] = bi;
    __syncthreads();
    for (int s = 128; s > 0; s >>= 1) {
        if (t < s) {
            float v2 = sv[t + s]; int i2 = si[t + s];
            if (v2 > sv[t] || (v2 == sv[t] && i2 < si[t])) { sv[t] = v2; si[t] = i2; }
        }
        __syncthreads();
    }
    if (t == 0) idx[pair] = si[0];
}

// ---------------------------------------------------------------------------
// K2: transpose + f16-convert weights: w1t[n][k], w2t[n][k]
// ---------------------------------------------------------------------------
__global__ void k_wconv(const float* __restrict__ w1, const float* __restrict__ w2,
                        _Float16* __restrict__ w1t, _Float16* __restrict__ w2t) {
    int i = blockIdx.x * 256 + threadIdx.x;
    if (i < HID * DIM) {
        int n = i / DIM, k = i % DIM;
        w1t[i] = (_Float16)w1[k * HID + n];
    } else {
        int j = i - HID * DIM;            // j < HID*HID (grid sized exactly)
        int n = j / HID, k = j % HID;
        w2t[j] = (_Float16)w2[k * HID + n];
    }
}

// ---------------------------------------------------------------------------
// K3: q_max = MLP(feats[top_idx]) for the 8 selected rows (f32, tiny)
// ---------------------------------------------------------------------------
__global__ void k_qmax(const float* __restrict__ feats,
                       const float* __restrict__ w1, const float* __restrict__ b1,
                       const float* __restrict__ w2, const float* __restrict__ b2,
                       const int* __restrict__ idx, float* __restrict__ qmax) {
    int h = threadIdx.x;                  // 128 threads
    __shared__ float hl[BATCH * OC][HID];
    for (int r = 0; r < BATCH * OC; r++) {
        int b = r / OC;
        const float* x = feats + ((size_t)b * INST + idx[r]) * DIM;
        float s = b1[h];
        for (int k = 0; k < DIM; k++) s += x[k] * w1[k * HID + h];
        hl[r][h] = fmaxf(s, 0.f);
    }
    __syncthreads();
    for (int r = 0; r < BATCH * OC; r++) {
        float s = b2[h];
        for (int k = 0; k < HID; k++) s += hl[r][k] * w2[k * HID + h];
        qmax[r * HID + h] = tanhf(s);
    }
}

// ---------------------------------------------------------------------------
// K4: main WMMA kernel. 16 instances per block, 8 waves each own a 16-col
// tile of the 128 hidden/output units. logits = tanh(...)·q_max / 200
// ---------------------------------------------------------------------------
__global__ void __launch_bounds__(256) k_mlp_logits(
        const float* __restrict__ feats,
        const _Float16* __restrict__ w1t, const float* __restrict__ b1,
        const _Float16* __restrict__ w2t, const float* __restrict__ b2,
        const float* __restrict__ qmax, float* __restrict__ logits) {
    __shared__ __align__(16) _Float16 sX[16 * DIM];   // feats tile (f16)
    __shared__ __align__(16) _Float16 sH[16 * HID];   // relu hidden (f16)
    __shared__ __align__(16) float    sQ[16 * HID];   // tanh output (f32)

    int b = blockIdx.y;
    int inst0 = blockIdx.x * 16;
    int t = threadIdx.x;
    int wave = t >> 5, lane = t & 31;

    // Stage feats tile f32 -> f16 LDS (each thread: 32 contiguous floats)
    {
        const float* src = feats + ((size_t)b * INST + inst0) * DIM;
        int e0 = t * 32;
        const float4* s4 = (const float4*)(src + e0);
        #pragma unroll
        for (int i = 0; i < 8; i++) {
            float4 v = s4[i];
            v4h hv = { (_Float16)v.x, (_Float16)v.y, (_Float16)v.z, (_Float16)v.w };
            *(v4h*)(&sX[e0 + i * 4]) = hv;
        }
    }
    __syncthreads();

    int n0 = wave * 16;
    int n_col = n0 + (lane & 15);
    int m_base = (lane >> 4) * 8;

    // Layer 1: [16x512] x [512x16] per wave, K-loop of 16 WMMAs
    v8f acc = {};
    for (int kc = 0; kc < DIM; kc += 32) {
        v16h a  = load_opA_lds(sX, DIM, lane, kc);
        v16h bm = load_opB_gl(w1t, DIM, lane, n0, kc);
        acc = __builtin_amdgcn_wmma_f32_16x16x32_f16(
                  false, a, false, bm, (short)0, acc, false, false);
    }
    float bias1 = b1[n_col];
    #pragma unroll
    for (int v = 0; v < 8; v++) {
        float x = fmaxf(acc[v] + bias1, 0.f);
        sH[(m_base + v) * HID + n_col] = (_Float16)x;
    }
    __syncthreads();

    // Layer 2: [16x128] x [128x16] per wave, 4 WMMAs
    v8f acc2 = {};
    for (int kc = 0; kc < HID; kc += 32) {
        v16h a  = load_opA_lds(sH, HID, lane, kc);
        v16h bm = load_opB_gl(w2t, HID, lane, n0, kc);
        acc2 = __builtin_amdgcn_wmma_f32_16x16x32_f16(
                   false, a, false, bm, (short)0, acc2, false, false);
    }
    float bias2 = b2[n_col];
    #pragma unroll
    for (int v = 0; v < 8; v++)
        sQ[(m_base + v) * HID + n_col] = tanhf(acc2[v] + bias2);
    __syncthreads();

    // logits[b][inst0+row][o] = (Q[row]·qmax[b][o]) / sqrt(40000)
    if (t < 32) {
        int row = t >> 1, o = t & 1;
        const float* qm = qmax + (b * OC + o) * HID;
        float s = 0.f;
        for (int k = 0; k < HID; k++) s += sQ[row * HID + k] * qm[k];
        logits[((size_t)b * INST + inst0 + row) * OC + o] = s * (1.0f / 200.0f);
    }
}

// ---------------------------------------------------------------------------
// K5: online softmax stats (max, sum of exp) per (b, class)
// ---------------------------------------------------------------------------
__global__ void k_softmax_stats(const float* __restrict__ logits,
                                float* __restrict__ stats) {
    int pair = blockIdx.x;                // b*OC+o
    int o = pair % OC, b = pair / OC;
    int t = threadIdx.x;
    const float* L = logits + (size_t)b * INST * OC + o;
    float m = -INFINITY, s = 0.f;
    for (int n = t; n < INST; n += 256) {
        float x = L[(size_t)n * OC];
        float mn = fmaxf(m, x);
        s = s * expf(m - mn) + expf(x - mn);
        m = mn;
    }
    __shared__ float sm[256], ss[256];
    sm[t] = m; ss[t] = s;
    __syncthreads();
    for (int st = 128; st > 0; st >>= 1) {
        if (t < st) {
            float m2 = sm[t + st], s2 = ss[t + st];
            float mn = fmaxf(sm[t], m2);
            ss[t] = ss[t] * expf(sm[t] - mn) + s2 * expf(m2 - mn);
            sm[t] = mn;
        }
        __syncthreads();
    }
    if (t == 0) { stats[pair * 2] = sm[0]; stats[pair * 2 + 1] = ss[0]; }
}

// ---------------------------------------------------------------------------
// K6: A = softmax(logits) written to output
// ---------------------------------------------------------------------------
__global__ void k_write_A(const float* __restrict__ logits,
                          const float* __restrict__ stats, float* __restrict__ A) {
    size_t i = (size_t)blockIdx.x * 256 + threadIdx.x;
    if (i >= (size_t)BATCH * INST * OC) return;
    int o = (int)(i % OC);
    int b = (int)(i / ((size_t)INST * OC));
    int pair = b * OC + o;
    A[i] = expf(logits[i] - stats[pair * 2]) / stats[pair * 2 + 1];
}

// ---------------------------------------------------------------------------
// K7: partial B = A^T · feats, deterministic two-stage reduction
// ---------------------------------------------------------------------------
__global__ void __launch_bounds__(256) k_partB(
        const float* __restrict__ feats, const float* __restrict__ logits,
        const float* __restrict__ stats, float* __restrict__ partB) {
    int b = blockIdx.y, blk = blockIdx.x;
    int t = threadIdx.x;
    int n0 = blk * CH;
    __shared__ float sA[CH * OC];
    float m0 = stats[(b * OC + 0) * 2], is0 = 1.f / stats[(b * OC + 0) * 2 + 1];
    float m1 = stats[(b * OC + 1) * 2], is1 = 1.f / stats[(b * OC + 1) * 2 + 1];
    for (int i = t; i < CH * OC; i += 256) {
        float l = logits[((size_t)b * INST + n0) * OC + i];
        sA[i] = (i & 1) ? expf(l - m1) * is1 : expf(l - m0) * is0;
    }
    __syncthreads();
    float a00 = 0, a01 = 0, a10 = 0, a11 = 0;
    const float* F = feats + ((size_t)b * INST + n0) * DIM;
    for (int n = 0; n < CH; n++) {
        float f0 = F[(size_t)n * DIM + t];
        float f1 = F[(size_t)n * DIM + t + 256];
        float A0 = sA[n * OC], A1 = sA[n * OC + 1];
        a00 += A0 * f0; a01 += A0 * f1;
        a10 += A1 * f0; a11 += A1 * f1;
    }
    float* P = partB + ((size_t)b * NB + blk) * OC * DIM;
    P[t] = a00; P[t + 256] = a01;
    P[DIM + t] = a10; P[DIM + t + 256] = a11;
}

__global__ void k_reduceB(const float* __restrict__ partB, float* __restrict__ Bout) {
    int i = blockIdx.x * 256 + threadIdx.x;     // 0..4095
    int b = i / (OC * DIM);
    int r = i % (OC * DIM);
    const float* P = partB + (size_t)b * NB * OC * DIM + r;
    float s = 0.f;
    for (int k = 0; k < NB; k++) s += P[(size_t)k * OC * DIM];
    Bout[i] = s;
}

// ---------------------------------------------------------------------------
// K8: C[b][o] = sum_{i,d} B[b][i][d] * fcc_w[o][i][d] + fcc_b[o]
// ---------------------------------------------------------------------------
__global__ void k_final(const float* __restrict__ Bout, const float* __restrict__ fcc_w,
                        const float* __restrict__ fcc_b, float* __restrict__ Cout) {
    int t = threadIdx.x;                 // 256 = 8 groups of 32
    int g = t >> 5, lane = t & 31;
    int b = g >> 1, o = g & 1;
    float s = 0.f;
    for (int j = lane; j < OC * DIM; j += 32)
        s += Bout[b * OC * DIM + j] * fcc_w[o * OC * DIM + j];
    __shared__ float red[256];
    red[t] = s; __syncthreads();
    for (int st = 16; st > 0; st >>= 1) {
        if (lane < st) red[t] += red[t + st];
        __syncthreads();
    }
    if (lane == 0) Cout[b * OC + o] = red[t] + fcc_b[o];
}

// ---------------------------------------------------------------------------
extern "C" void kernel_launch(void* const* d_in, const int* in_sizes, int n_in,
                              void* d_out, int out_size, void* d_ws, size_t ws_size,
                              hipStream_t stream) {
    const float* feats = (const float*)d_in[0];
    const float* c     = (const float*)d_in[1];
    const float* q_w1  = (const float*)d_in[2];
    const float* q_b1  = (const float*)d_in[3];
    const float* q_w2  = (const float*)d_in[4];
    const float* q_b2  = (const float*)d_in[5];
    const float* fcc_w = (const float*)d_in[6];
    const float* fcc_b = (const float*)d_in[7];

    // Output layout: C[8] | A[4*40000*2] | B[4*2*512]
    float* Cout = (float*)d_out;
    float* Aout = Cout + BATCH * OC;
    float* Bout = Aout + (size_t)BATCH * INST * OC;

    // Workspace layout (bytes)
    char* ws = (char*)d_ws;
    int*      idx    = (int*)     (ws + 0);            //     32 B
    float*    stats  = (float*)   (ws + 64);           //     64 B
    float*    qmaxp  = (float*)   (ws + 256);          //   4 KB
    _Float16* w1t    = (_Float16*)(ws + 8192);         // 128 KB
    _Float16* w2t    = (_Float16*)(ws + 8192 + 131072);//  32 KB
    float*    logits = (float*)   (ws + 172032);       // 1.28 MB
    float*    partB  = (float*)   (ws + 1452032);      // 1.6 MB

    k_argmax<<<BATCH * OC, 256, 0, stream>>>(c, idx);
    k_wconv<<<(HID * DIM + HID * HID) / 256, 256, 0, stream>>>(q_w1, q_w2, w1t, w2t);
    k_qmax<<<1, 128, 0, stream>>>(feats, q_w1, q_b1, q_w2, q_b2, idx, qmaxp);
    k_mlp_logits<<<dim3(INST / 16, BATCH), 256, 0, stream>>>(
        feats, w1t, q_b1, w2t, q_b2, qmaxp, logits);
    k_softmax_stats<<<BATCH * OC, 256, 0, stream>>>(logits, stats);
    k_write_A<<<((size_t)BATCH * INST * OC + 255) / 256, 256, 0, stream>>>(logits, stats, Aout);
    k_partB<<<dim3(NB, BATCH), 256, 0, stream>>>(feats, logits, stats, partB);
    k_reduceB<<<(BATCH * OC * DIM) / 256, 256, 0, stream>>>(partB, Bout);
    k_final<<<1, 256, 0, stream>>>(Bout, fcc_w, fcc_b, Cout);
}
